// Decoder_85066122265358
// MI455X (gfx1250) — matrix operand order repaired and checked by reading, not compiled, not run on hardware
//
#include <hip/hip_runtime.h>
#include <math.h>

typedef __attribute__((ext_vector_type(16))) _Float16 v16h;
typedef __attribute__((ext_vector_type(8)))  float    v8f;

#define S_  128
#define P_  32
#define HD_ 32
#define ED_ 16
#define GH_ 72
#define GO_ 8
#define MD_ 64
#define T_  8
#define B_  (S_ * P_)
#define NTHREADS 128
#define NWAVES   4
#define BLD      80   // leading dim (f16 elems) of row-major per-wave buffers
#define XD_      (ED_ + HD_)   // 48: GCN input feature dim

// ---- shared memory layout (bytes) ----
//  hS    float[32][32]    @     0  (4096)
//  posS  float[32][2]     @  4096  ( 256)
//  xsrcS float[32][2]     @  4352  ( 256)
//  gS    int[32]          @  4608  ( 128)
//  phS   float[32][16]    @  4736  (2048)
//  wavebuf (4 waves)      @  6784  (4 * 12288 = 49152)
//     per wave: aBuf f16[32][32] (+0) | bufA f16[80][32] Y1T (+2048) | bufB f16[32][80] (+7168)
//  w1Lds f16[72][48] [N][K] @ 55936 (6912)   staged per (t,type)
//  w2Lds f16[8][72]  [N][K] @ 62848 (1152)   staged per (t,type)
//  total = 64000 bytes
#define SMEM_BYTES 64000

static __device__ __forceinline__ void wave_lds_fence() {
  __builtin_amdgcn_wave_barrier();
  asm volatile("s_wait_dscnt 0" ::: "memory");
  __builtin_amdgcn_wave_barrier();
}

static __device__ __forceinline__ float sigm(float x) { return 1.0f / (1.0f + expf(-x)); }

// A fragment: 16x32 f16 tile from row-major f16 LDS (stride ld), zero-pad past kmax.
// Per-lane contiguous -> ds_load_b128 pairs.
static __device__ __forceinline__ v16h frag_a16(const _Float16* src, int ld, int row0, int col0,
                                                int kmax, int lane) {
  int m  = row0 + (lane & 15);
  int hb = ((lane >> 4) & 1) << 4;
  v16h a;
#pragma unroll
  for (int e = 0; e < 16; ++e) {
    int k = col0 + hb + e;
    a[e]  = (k < kmax) ? src[m * ld + k] : (_Float16)0.0f;
  }
  return a;
}
// B fragment from a TRANSPOSED f16 LDS matrix stored [N][K] (stride ldT), K-tile of 32 rows.
// Per-lane contiguous 32B -> 2x ds_load_b128; guarded zero-padding past kmax/nmax.
static __device__ __forceinline__ v16h frag_bT16(const _Float16* srcT, int ldT, int row0,
                                                 int col0, int kmax, int nmax, int lane) {
  int n  = col0 + (lane & 15);
  int kb = ((lane >> 4) & 1) << 4;
  v16h b;
#pragma unroll
  for (int e = 0; e < 16; ++e) {
    int k = row0 + kb + e;
    b[e]  = (k < kmax && n < nmax) ? srcT[n * ldT + k] : (_Float16)0.0f;
  }
  return b;
}
// Build the spatial-embedding B fragment of X (cols 0..15) directly in registers.
// lane -> column n (embedding feature); rows k are pedestrians.
static __device__ __forceinline__ v16h xfrag_emb(const float* posS, int i,
                                                 const float* __restrict__ W_pse,
                                                 const float* __restrict__ b_pse, int lane) {
  int n  = lane & 15;
  int kb = ((lane >> 4) & 1) << 4;
  float w0 = W_pse[n], w1 = W_pse[ED_ + n], bb = b_pse[n];
  float pix = posS[2 * i], piy = posS[2 * i + 1];
  v16h b;
#pragma unroll
  for (int e = 0; e < 16; ++e) {
    int k   = kb + e;
    float ex = posS[2 * k] - pix;     // rel[i,k] = pos[k] - pos[i]
    float ey = posS[2 * k + 1] - piy;
    b[e] = (_Float16)(ex * w0 + ey * w1 + bb);
  }
  return b;
}
// Build a hidden-state B fragment of X (cols c0..c0+15 of h) directly in registers.
static __device__ __forceinline__ v16h xfrag_h(const float* hS, int c0, int lane) {
  int n  = (lane & 15) + c0;
  int kb = ((lane >> 4) & 1) << 4;
  v16h b;
#pragma unroll
  for (int e = 0; e < 16; ++e) b[e] = (_Float16)hS[(kb + e) * HD_ + n];
  return b;
}
// Store f32 C fragment to row-major f16 LDS tile (strided; feeds later A-operands).
static __device__ __forceinline__ void store_c16(_Float16* dst, int ld, int row0, int col0,
                                                 v8f c, int lane) {
  int n  = col0 + (lane & 15);
  int mb = row0 + (((lane >> 4) & 1) << 3);
#pragma unroll
  for (int r = 0; r < 8; ++r) dst[(mb + r) * ld + n] = (_Float16)c[r];
}
// Store f32 C fragment TRANSPOSED to f16 LDS stored [N][M] (stride ldT), with ReLU.
// Per-lane contiguous 16B -> single ds_store_b128 per tile.
static __device__ __forceinline__ void store_cT16(_Float16* dstT, int ldT, int row0, int col0,
                                                  v8f c, int lane) {
  int n  = col0 + (lane & 15);
  int mb = row0 + (((lane >> 4) & 1) << 3);
#pragma unroll
  for (int r = 0; r < 8; ++r) dstT[n * ldT + mb + r] = (_Float16)fmaxf(c[r], 0.0f);
}

#define WMMA(a, b, c) \
  __builtin_amdgcn_wmma_f32_16x16x32_f16(false, (a), false, (b), (short)0, (c), false, false)

__global__ __launch_bounds__(NTHREADS, 1) void decoder_scene_kernel(
    const float* __restrict__ last_pos, const float* __restrict__ last_pos_rel,
    const float* __restrict__ hh, const float* __restrict__ ch,
    const int* __restrict__ end_group,
    const float* __restrict__ W_se, const float* __restrict__ b_se,
    const float* __restrict__ Wih, const float* __restrict__ Whh,
    const float* __restrict__ bih, const float* __restrict__ bhh,
    const float* __restrict__ W_hp, const float* __restrict__ b_hp,
    const float* __restrict__ W_pse, const float* __restrict__ b_pse,
    const float* __restrict__ W1a, const float* __restrict__ W2a,
    const float* __restrict__ W1b, const float* __restrict__ W2b,
    const float* __restrict__ W_m1, const float* __restrict__ b_m1,
    const float* __restrict__ W_m2, const float* __restrict__ b_m2,
    float* __restrict__ out) {
  __shared__ __align__(16) unsigned char smem[SMEM_BYTES];
  float* hS     = (float*)(smem + 0);
  float* posS   = (float*)(smem + 4096);
  float* xsrcS  = (float*)(smem + 4352);
  int*   gS     = (int*)(smem + 4608);
  float* phS    = (float*)(smem + 4736);
  unsigned char* wb = smem + 6784;
  _Float16* w1Lds = (_Float16*)(smem + 55936);  // [72][48] = W1^T, B-fast layout
  _Float16* w2Lds = (_Float16*)(smem + 62848);  // [8][72]  = W2^T, B-fast layout

  const int tid  = threadIdx.x;
  const int s    = blockIdx.x;
  const int lane = tid & 31;
  const int w    = tid >> 5;

  _Float16* aBuf = (_Float16*)(wb + w * 12288);          // adjacency, 32x32 row-major
  _Float16* bufA = (_Float16*)(wb + w * 12288 + 2048);   // Y1T, 80x32 (stored [N][M])
  _Float16* bufB = (_Float16*)(wb + w * 12288 + 7168);   // M1 / M2, 32x80 row-major
  float*    dhS  = (float*)wb;  // 32*64 f32 = 8KB, aliases wave buffers (MLP phase only)

  const int myPed = tid >> 2;        // 0..31
  const int myU0  = (tid & 3) * 8;   // 0,8,16,24

  // ---- load initial state ----
  for (int e = tid; e < P_ * HD_; e += NTHREADS) hS[e] = hh[s * P_ * HD_ + e];
  if (tid < P_ * 2) {
    posS[tid]  = last_pos[s * P_ * 2 + tid];
    xsrcS[tid] = last_pos_rel[s * P_ * 2 + tid];
  }
  if (tid < P_) gS[tid] = end_group[s * P_ + tid];
  float creg[8];
#pragma unroll
  for (int j = 0; j < 8; ++j) creg[j] = ch[(s * P_ + myPed) * HD_ + myU0 + j];
  __syncthreads();

  for (int t = 0; t < T_; ++t) {
    // ================= Phase 1: LSTM cell (VALU) =================
    float h2loc[8];
    {
      float xk[ED_];
      float rx = xsrcS[myPed * 2 + 0], ry = xsrcS[myPed * 2 + 1];
#pragma unroll
      for (int k = 0; k < ED_; ++k)
        xk[k] = rx * W_se[k] + ry * W_se[ED_ + k] + b_se[k];
#pragma unroll
      for (int j = 0; j < 8; ++j) {
        int u = myU0 + j;
        float gi = bih[u] + bhh[u];
        float gf = bih[HD_ + u] + bhh[HD_ + u];
        float gg = bih[2 * HD_ + u] + bhh[2 * HD_ + u];
        float go = bih[3 * HD_ + u] + bhh[3 * HD_ + u];
        for (int k = 0; k < ED_; ++k) {
          float x = xk[k];
          gi += x * Wih[u * ED_ + k];
          gf += x * Wih[(HD_ + u) * ED_ + k];
          gg += x * Wih[(2 * HD_ + u) * ED_ + k];
          go += x * Wih[(3 * HD_ + u) * ED_ + k];
        }
        for (int k = 0; k < HD_; ++k) {
          float hk = hS[myPed * HD_ + k];
          gi += hk * Whh[u * HD_ + k];
          gf += hk * Whh[(HD_ + u) * HD_ + k];
          gg += hk * Whh[(2 * HD_ + u) * HD_ + k];
          go += hk * Whh[(3 * HD_ + u) * HD_ + k];
        }
        float c2 = sigm(gf) * creg[j] + sigm(gi) * tanhf(gg);
        float h2 = sigm(go) * tanhf(c2);
        creg[j]  = c2;
        h2loc[j] = h2;
      }
    }
#pragma unroll
    for (int j = 0; j < 8; ++j) hS[myPed * HD_ + myU0 + j] = h2loc[j];
    __syncthreads();

    // ================= Phase 2: hidden2pos, output, state update =================
    if (tid < P_ * 2) {
      int p = tid >> 1, d = tid & 1;
      float acc = b_hp[d];
      for (int k = 0; k < HD_; ++k) acc += hS[p * HD_ + k] * W_hp[k * 2 + d];
      out[((size_t)t * B_ + s * P_ + p) * 2 + d] = acc;   // rel_pos
      posS[tid] += acc;                                   // cur = rel + last
      xsrcS[tid] = acc;                                   // next LSTM input source
    }
    __syncthreads();

    // ================= Phase 3: pooling GCNs (WMMA, wave-private) =================
    for (int type = 0; type < 2; ++type) {
      const float* W1 = type ? W1b : W1a;
      const float* W2 = type ? W2b : W2a;
      // Cooperative stage: W^T in f16, B-fast [N][K] layout, shared by all waves.
      __syncthreads();  // previous type's weights no longer in use
      for (int e = tid; e < GH_ * XD_; e += NTHREADS) {   // W1 (48x72): e = k*72+n
        int n = e % GH_, k = e / GH_;
        w1Lds[n * XD_ + k] = (_Float16)W1[e];
      }
      for (int e = tid; e < GH_ * GO_; e += NTHREADS) {   // W2 (72x8): e = k*8+n
        int n = e & 7, k = e >> 3;
        w2Lds[n * GH_ + k] = (_Float16)W2[e];
      }
      __syncthreads();

      // W2 fragments are tiny (24 VGPRs): keep register-resident per type.
      v16h w2f[3];
#pragma unroll
      for (int kt = 0; kt < 3; ++kt)
        w2f[kt] = frag_bT16(w2Lds, GH_, kt * 32, 0, GH_, GO_, lane);

      for (int i = w; i < P_; i += NWAVES) {
        // --- build normalized A_i (32x32 f16) in aBuf; lane = row j ---
        {
          int j  = lane;
          int gi = gS[i], gj = gS[j];
          float denom = 0.0f;
#pragma unroll
          for (int k = 0; k < P_; ++k) {  // denom[j,i] = sum_k adj(j,i,k)*mask(i,k)
            bool adjv = (i == k) || (j == i) || (j == k);
            bool sc   = (gi == gS[k]) && (gi != 0);
            bool m    = (type == 0) ? (sc || i == k) : (!sc || i == k);
            denom += (adjv && m) ? 1.0f : 0.0f;
          }
          float inv = 1.0f / denom;
#pragma unroll
          for (int k = 0; k < P_; ++k) {  // A[i][j][k] = adj(i,j,k)*mask(j,k) / denom[j,i]
            bool adjv = (j == k) || (i == j) || (i == k);
            bool sc   = (gj == gS[k]) && (gj != 0);
            bool m    = (type == 0) ? (sc || j == k) : (!sc || j == k);
            aBuf[j * P_ + k] = (_Float16)((adjv && m) ? inv : 0.0f);
          }
        }
        wave_lds_fence();
        // Adjacency A-fragments, reused by GEMM1 and GEMM3.
        v16h af0 = frag_a16(aBuf, P_, 0, 0, P_, lane);
        v16h af1 = frag_a16(aBuf, P_, 16, 0, P_, lane);

        // GEMM1: M1 = A(32x32) @ X(32x48) -> bufB row-major. X fragments built in registers.
        {
          v16h b = xfrag_emb(posS, i, W_pse, b_pse, lane);
          v8f c0 = {}; c0 = WMMA(af0, b, c0); store_c16(bufB, BLD, 0, 0, c0, lane);
          v8f c1 = {}; c1 = WMMA(af1, b, c1); store_c16(bufB, BLD, 16, 0, c1, lane);
          b = xfrag_h(hS, 0, lane);
          v8f c2 = {}; c2 = WMMA(af0, b, c2); store_c16(bufB, BLD, 0, 16, c2, lane);
          v8f c3 = {}; c3 = WMMA(af1, b, c3); store_c16(bufB, BLD, 16, 16, c3, lane);
          b = xfrag_h(hS, 16, lane);
          v8f c4 = {}; c4 = WMMA(af0, b, c4); store_c16(bufB, BLD, 0, 32, c4, lane);
          v8f c5 = {}; c5 = WMMA(af1, b, c5); store_c16(bufB, BLD, 16, 32, c5, lane);
        }
        wave_lds_fence();

        // GEMM2: Y1 = relu(M1(32x48) @ W1(48x72)) stored TRANSPOSED -> bufA = Y1T[80][32]
        {
          v16h a2[2][2];
#pragma unroll
          for (int mt = 0; mt < 2; ++mt)
#pragma unroll
            for (int kt = 0; kt < 2; ++kt)
              a2[mt][kt] = frag_a16(bufB, BLD, mt * 16, kt * 32, XD_, lane);
#pragma unroll
          for (int nt = 0; nt < 5; ++nt) {
            v16h w10 = frag_bT16(w1Lds, XD_, 0, nt * 16, XD_, GH_, lane);
            v16h w11 = frag_bT16(w1Lds, XD_, 32, nt * 16, XD_, GH_, lane);
#pragma unroll
            for (int mt = 0; mt < 2; ++mt) {
              v8f acc = {};
              acc = WMMA(a2[mt][0], w10, acc);
              acc = WMMA(a2[mt][1], w11, acc);
              store_cT16(bufA, P_, mt * 16, nt * 16, acc, lane);   // ReLU inside
            }
          }
        }
        wave_lds_fence();

        // GEMM3: M2 = A(32x32) @ Y1(32x80) -> bufB row-major. B-frags contiguous from Y1T.
#pragma unroll
        for (int nt = 0; nt < 5; ++nt) {
          v16h b = frag_bT16(bufA, P_, 0, nt * 16, P_, BLD, lane);
          v8f c0 = {}; c0 = WMMA(af0, b, c0); store_c16(bufB, BLD, 0, nt * 16, c0, lane);
          v8f c1 = {}; c1 = WMMA(af1, b, c1); store_c16(bufB, BLD, 16, nt * 16, c1, lane);
        }
        wave_lds_fence();

        // GEMM4: Y2 = relu(M2(32x80) @ W2(72x8 padded)); fused max over rows j.
        float redmax = -3.4e38f;
#pragma unroll
        for (int mt = 0; mt < 2; ++mt) {
          v8f acc = {};
#pragma unroll
          for (int kt = 0; kt < 3; ++kt) {
            v16h a = frag_a16(bufB, BLD, mt * 16, kt * 32, BLD, lane);
            acc    = WMMA(a, w2f[kt], acc);
          }
#pragma unroll
          for (int r = 0; r < 8; ++r) redmax = fmaxf(redmax, fmaxf(acc[r], 0.0f));
        }
        float omax = fmaxf(redmax, __shfl_xor(redmax, 16, 32));
        if (lane < GO_) phS[i * 16 + type * GO_ + lane] = omax;
      }
    }
    __syncthreads();

    // ================= Phase 4: MLP -> new hidden state =================
    for (int e = tid; e < P_ * MD_; e += NTHREADS) {
      int p = e >> 6, m = e & 63;
      float acc = b_m1[m];
      for (int k = 0; k < HD_; ++k) acc += hS[p * HD_ + k] * W_m1[k * MD_ + m];
      for (int k = 0; k < 2 * GO_; ++k) acc += phS[p * 16 + k] * W_m1[(HD_ + k) * MD_ + m];
      dhS[e] = fmaxf(acc, 0.0f);
    }
    __syncthreads();
    for (int e = tid; e < P_ * HD_; e += NTHREADS) {
      int p = e >> 5, u = e & 31;
      float acc = b_m2[u];
      for (int k = 0; k < MD_; ++k) acc += dhS[p * MD_ + k] * W_m2[k * HD_ + u];
      hS[e] = fmaxf(acc, 0.0f);
    }
    __syncthreads();
  }
}

extern "C" void kernel_launch(void* const* d_in, const int* in_sizes, int n_in,
                              void* d_out, int out_size, void* d_ws, size_t ws_size,
                              hipStream_t stream) {
  (void)in_sizes; (void)n_in; (void)out_size; (void)d_ws; (void)ws_size;
  const float* last_pos     = (const float*)d_in[0];
  const float* last_pos_rel = (const float*)d_in[1];
  const float* hh           = (const float*)d_in[2];
  const float* ch           = (const float*)d_in[3];
  /* d_in[4] = seq_start_end (int64) — scenes are uniform [s*P,(s+1)*P); unused */
  const int*   end_group    = (const int*)d_in[5];
  const float* W_se  = (const float*)d_in[6];
  const float* b_se  = (const float*)d_in[7];
  const float* Wih   = (const float*)d_in[8];
  const float* Whh   = (const float*)d_in[9];
  const float* bih   = (const float*)d_in[10];
  const float* bhh   = (const float*)d_in[11];
  const float* W_hp  = (const float*)d_in[12];
  const float* b_hp  = (const float*)d_in[13];
  const float* W_pse = (const float*)d_in[14];
  const float* b_pse = (const float*)d_in[15];
  const float* W1a   = (const float*)d_in[16];
  const float* W2a   = (const float*)d_in[17];
  const float* W1b   = (const float*)d_in[18];
  const float* W2b   = (const float*)d_in[19];
  const float* W_m1  = (const float*)d_in[20];
  const float* b_m1  = (const float*)d_in[21];
  const float* W_m2  = (const float*)d_in[22];
  const float* b_m2  = (const float*)d_in[23];
  float* out = (float*)d_out;

  decoder_scene_kernel<<<dim3(S_), dim3(NTHREADS), 0, stream>>>(
      last_pos, last_pos_rel, hh, ch, end_group,
      W_se, b_se, Wih, Whh, bih, bhh, W_hp, b_hp, W_pse, b_pse,
      W1a, W2a, W1b, W2b, W_m1, b_m1, W_m2, b_m2, out);
}
